// NeighAttnMat_16398185136398
// MI455X (gfx1250) — matrix operand order repaired and checked by reading, not compiled, not run on hardware
//
#include <hip/hip_runtime.h>
#include <hip/hip_bf16.h>

// ---------------- problem constants ----------------
#define BB    8
#define HH    160
#define WW    160
#define CC    128
#define HEADS 4
#define HD    32
#define KK    7
#define RAD   3
#define TAPS  49
#define HALO  22            // 16 + (KK-1), covers clamped window at borders
#define MTOT  (BB*HH*WW)    // 204800 rows of the projection GEMM
#define NTOT  256           // 2*C output features
#define KTOT  128           // C input features

typedef __attribute__((ext_vector_type(16))) _Float16     v16h;
typedef __attribute__((ext_vector_type(8)))  float        v8f;
typedef __attribute__((ext_vector_type(4)))  int          v4i;
typedef __attribute__((ext_vector_type(8)))  int          v8i;
typedef __attribute__((ext_vector_type(4)))  unsigned int v4u;

#define AS1 __attribute__((address_space(1)))
#define AS3 __attribute__((address_space(3)))

// --------- CDNA5 async global->LDS + TDM paths (guarded; fall back to sync copy) ---------
#if defined(__has_builtin)
# if __has_builtin(__builtin_amdgcn_global_load_async_to_lds_b128)
#  define HAVE_ASYNC_B128 1
# endif
# if __has_builtin(__builtin_amdgcn_s_wait_asynccnt)
#  define HAVE_WAIT_ASYNC 1
# endif
# if __has_builtin(__builtin_amdgcn_tensor_load_to_lds)
#  define HAVE_TDM 1
# endif
# if __has_builtin(__builtin_amdgcn_s_wait_tensorcnt)
#  define HAVE_WAIT_TENSOR 1
# endif
#endif

__device__ __forceinline__ void g2lds_b128(void* lds_dst, const void* gsrc) {
#if defined(HAVE_ASYNC_B128)
    // Param types per hipcc diagnostic: (v4i AS1*, v4i AS3*, imm int, imm int).
    __builtin_amdgcn_global_load_async_to_lds_b128(
        (AS1 v4i*)(unsigned long long)(size_t)gsrc,
        (AS3 v4i*)(unsigned)(size_t)lds_dst,
        0, 0);
#else
    *(float4*)lds_dst = *(const float4*)gsrc;
#endif
}

__device__ __forceinline__ void g2lds_wait() {
#if defined(HAVE_ASYNC_B128)
# if defined(HAVE_WAIT_ASYNC)
    __builtin_amdgcn_s_wait_asynccnt(0);
# else
    asm volatile("s_wait_asynccnt 0x0" ::: "memory");
# endif
#endif
}

__device__ __forceinline__ void tensor_wait() {
#if defined(HAVE_WAIT_TENSOR)
    __builtin_amdgcn_s_wait_tensorcnt(0);
#else
    asm volatile("s_wait_tensorcnt 0x0" ::: "memory");
#endif
}

// ---------------- kernel 0: fp32 -> f16 weight conversion ----------------
__global__ __launch_bounds__(256) void convert_w_kernel(const float* __restrict__ w,
                                                        _Float16* __restrict__ wh, int n) {
    int i = blockIdx.x * blockDim.x + threadIdx.x;
    if (i < n) wh[i] = (_Float16)w[i];
}

// ---------------- kernel 1: qk = x @ W^T + b via WMMA f16->f32 ----------------
// W (f16, 64KB) is async-staged into LDS once per block. Each of the 8 waves owns a
// 16-row M tile and sweeps all 16 N tiles with register double-buffered B fragments.
__device__ __forceinline__ void load_b_frags(v16h* dst, const _Float16* sW,
                                             int nt, int col, int khalf) {
    const _Float16* wr = sW + (size_t)(nt * 16 + col) * KTOT + khalf * 16;
#pragma unroll
    for (int kf = 0; kf < 4; ++kf)
        dst[kf] = *(const v16h*)(wr + kf * 32);
}

__global__ __launch_bounds__(256) void qk_proj_kernel(const float* __restrict__ x,
                                                      const _Float16* __restrict__ wh,
                                                      const float* __restrict__ bias,
                                                      float* __restrict__ qk) {
    __shared__ __align__(32) _Float16 sW[NTOT * KTOT];   // 64 KB

    // ---- async stage of W into LDS (4096 x b128, 16 per thread) ----
    for (int idx = threadIdx.x; idx < (NTOT * KTOT) / 8; idx += 256)
        g2lds_b128(sW + (size_t)idx * 8, wh + (size_t)idx * 8);

    const int lane  = threadIdx.x & 31;
    const int wave  = threadIdx.x >> 5;
    const int m0    = blockIdx.x * 128 + wave * 16;
    const int khalf = lane >> 4;
    const int col   = lane & 15;

    // ---- build 4 A fragments (overlaps with async W staging) ----
    const float* xrow = x + (size_t)(m0 + col) * KTOT;
    v16h a[4];
#pragma unroll
    for (int kf = 0; kf < 4; ++kf) {
        const int off = kf * 32 + khalf * 16;
        float4 t0 = *(const float4*)(xrow + off);
        float4 t1 = *(const float4*)(xrow + off + 4);
        float4 t2 = *(const float4*)(xrow + off + 8);
        float4 t3 = *(const float4*)(xrow + off + 12);
        v16h af;
        af[0]  = (_Float16)t0.x; af[1]  = (_Float16)t0.y; af[2]  = (_Float16)t0.z; af[3]  = (_Float16)t0.w;
        af[4]  = (_Float16)t1.x; af[5]  = (_Float16)t1.y; af[6]  = (_Float16)t1.z; af[7]  = (_Float16)t1.w;
        af[8]  = (_Float16)t2.x; af[9]  = (_Float16)t2.y; af[10] = (_Float16)t2.z; af[11] = (_Float16)t2.w;
        af[12] = (_Float16)t3.x; af[13] = (_Float16)t3.y; af[14] = (_Float16)t3.z; af[15] = (_Float16)t3.w;
        a[kf] = af;
    }

    g2lds_wait();
    __syncthreads();

    // ---- sweep the 16 N tiles with double-buffered B fragments ----
    v16h bufA[4], bufB[4];
    load_b_frags(bufA, sW, 0, col, khalf);

#pragma unroll
    for (int nt = 0; nt < 16; ++nt) {
        v16h* cur = (nt & 1) ? bufB : bufA;
        v16h* nxt = (nt & 1) ? bufA : bufB;
        if (nt < 15) load_b_frags(nxt, sW, nt + 1, col, khalf);

        const int n  = nt * 16 + col;
        const float bv = bias[n];
        v8f c;
#pragma unroll
        for (int r = 0; r < 8; ++r) c[r] = bv;

#pragma unroll
        for (int kf = 0; kf < 4; ++kf)
            c = __builtin_amdgcn_wmma_f32_16x16x32_f16(false, a[kf], false, cur[kf],
                                                       (short)0, c, false, false);

        float* outp = qk + (size_t)(m0 + khalf * 8) * NTOT + n;
#pragma unroll
        for (int r = 0; r < 8; ++r) outp[(size_t)r * NTOT] = c[r];
    }
}

// ---------------- kernel 2: neighborhood attention scores ----------------
// k-halo (32 floats x 22 px x 22 rows, strided) staged into LDS by the Tensor Data
// Mover: one TENSOR_LOAD_TO_LDS per block instead of ~3872 per-lane vector loads.
__global__ __launch_bounds__(256) void nattn_kernel(const float* __restrict__ qk,
                                                    const float* __restrict__ rpb,
                                                    float* __restrict__ out) {
    extern __shared__ float sK[];            // HALO*HALO pixels * 32 floats = 60.5 KB
    const int bh = blockIdx.y;               // b*HEADS + h
    const int b  = bh >> 2;
    const int h  = bh & 3;
    const int i0 = (blockIdx.x / 10) * 16;
    const int j0 = (blockIdx.x % 10) * 16;

    const int ri0 = min(max(i0 - RAD, 0), HH - KK);   // top-left of halo region
    const int rj0 = min(max(j0 - RAD, 0), WW - KK);

    const size_t kbase = (size_t)b * (HH * WW) * NTOT + 128 + h * HD;

#if defined(HAVE_TDM)
    if (threadIdx.x < 32) {        // wave 0 issues the DMA (EXEC is ignored by TDM)
        const unsigned long long gaddr =
            (unsigned long long)(size_t)(qk + kbase + ((size_t)ri0 * WW + rj0) * NTOT);
        const unsigned ldsaddr = (unsigned)(size_t)sK;   // flat low 32 = LDS offset

        // ---- D# group 0: count=1, lds_addr, 57-bit global_addr, type=2 ----
        v4u g0;
        g0[0] = 1u;
        g0[1] = ldsaddr;
        g0[2] = (unsigned)(gaddr & 0xFFFFFFFFu);
        g0[3] = (unsigned)((gaddr >> 32) & 0x01FFFFFFu) | (2u << 30);

        // ---- D# group 1: data_size=4B; dims/strides in element units ----
        // tile: dim0=32 (f32 head), dim1=22 px, dim2=22 rows
        // tensor (remaining extent from tile start): dim0=32, dim1=WW-rj0
        // stride0 = 256 (between pixels), stride1 = WW*256 (between rows)
        const unsigned td1 = (unsigned)(WW - rj0);
        const unsigned s1  = (unsigned)(WW * NTOT);      // 40960
        v8i g1;
        g1[0] = (int)(2u << 16);                         // data_size = 4 bytes
        g1[1] = (int)(32u << 16);                        // tensor_dim0[15:0] @ bits63:48
        g1[2] = (int)((td1 & 0xFFFFu) << 16);            // tensor_dim1[15:0] @ bits95:80
        g1[3] = (int)((td1 >> 16) | (32u << 16));        // tensor_dim1 hi | tile_dim0=32
        g1[4] = (int)(22u | (22u << 16));                // tile_dim1 | tile_dim2
        g1[5] = (int)(unsigned)NTOT;                     // tensor_dim0_stride[31:0]
        g1[6] = (int)((s1 & 0xFFFFu) << 16);             // stride0 hi=0 | stride1[15:0]
        g1[7] = (int)(s1 >> 16);                         // stride1[47:16]

        // ---- D# group 2: tensor_dim2 (remaining rows); dim3 unused ----
        v4i g2;
        g2[0] = HH - ri0; g2[1] = 0; g2[2] = 0; g2[3] = 0;
        v4i g3 = {0, 0, 0, 0};
        v8i g4 = {0, 0, 0, 0, 0, 0, 0, 0};   // extra group of the 6-arg builtin (unused)

        __builtin_amdgcn_tensor_load_to_lds(g0, g1, g2, g3, g4, 0);
        tensor_wait();
    }
#else
    for (int idx = threadIdx.x; idx < HALO * HALO * 8; idx += 256) {
        const int px = idx >> 3, c4 = idx & 7;
        int gi = ri0 + px / HALO; if (gi > HH - 1) gi = HH - 1;
        int gj = rj0 + px % HALO; if (gj > WW - 1) gj = WW - 1;
        g2lds_b128(sK + px * HD + c4 * 4,
                   qk + kbase + ((size_t)gi * WW + gj) * NTOT + c4 * 4);
    }
#endif

    const int li = threadIdx.x >> 4, lj = threadIdx.x & 15;
    const int i = i0 + li, j = j0 + lj;

    // q loads are LOADcnt-tracked and independent of the staging — issue now
    const float* qp = qk + ((size_t)b * (HH * WW) + (size_t)i * WW + j) * NTOT + h * HD;
    float4 qv[8];
#pragma unroll
    for (int t = 0; t < 8; ++t) qv[t] = *(const float4*)(qp + t * 4);

    const int nii = min(max(i - RAD, 0), HH - KK);
    const int njj = min(max(j - RAD, 0), WW - KK);
    const int lri = nii - ri0, lrj = njj - rj0;
    const int pi  = nii - i + (KK - 1);
    const int pj  = njj - j + (KK - 1);

#if !defined(HAVE_TDM)
    g2lds_wait();
#endif
    __syncthreads();

    float* op = out + (((size_t)bh * HH + i) * WW + j) * TAPS;
    const float* rb = rpb + h * (2 * KK - 1) * (2 * KK - 1);

    for (int ki = 0; ki < KK; ++ki) {
#pragma unroll
        for (int kj = 0; kj < KK; ++kj) {
            const float* kv = sK + ((lri + ki) * HALO + (lrj + kj)) * HD;
            float acc = rb[(pi + ki) * (2 * KK - 1) + (pj + kj)];
#pragma unroll
            for (int t = 0; t < 8; ++t) {
                const float4 k4 = *(const float4*)(kv + t * 4);
                acc += qv[t].x * k4.x + qv[t].y * k4.y + qv[t].z * k4.z + qv[t].w * k4.w;
            }
            op[ki * KK + kj] = acc;
        }
    }
}

// ---------------- host launcher ----------------
extern "C" void kernel_launch(void* const* d_in, const int* in_sizes, int n_in,
                              void* d_out, int out_size, void* d_ws, size_t ws_size,
                              hipStream_t stream) {
    const float* x    = (const float*)d_in[0];   // (8,160,160,128)
    const float* qk_w = (const float*)d_in[1];   // (256,128)
    const float* qk_b = (const float*)d_in[2];   // (256,)
    const float* rpb  = (const float*)d_in[3];   // (4,13,13)
    float*       out  = (float*)d_out;           // (8,4,160,160,49)

    // workspace layout: [qk fp32 buffer | f16 weights]
    const size_t QK_BYTES = (size_t)MTOT * NTOT * sizeof(float);   // 209,715,200 B
    float*     qk = (float*)d_ws;
    _Float16*  wh = (_Float16*)((char*)d_ws + QK_BYTES);

    // 0) convert weights to f16
    convert_w_kernel<<<dim3((NTOT * KTOT + 255) / 256), dim3(256), 0, stream>>>(
        qk_w, wh, NTOT * KTOT);

    // 1) projection GEMM with WMMA (8 waves/block, 16 M-rows per wave)
    qk_proj_kernel<<<dim3(MTOT / 128), dim3(256), 0, stream>>>(x, wh, qk_b, qk);

    // 2) neighborhood attention scores (TDM-staged k halo)
    const size_t smem = (size_t)HALO * HALO * HD * sizeof(float);  // 61,952 B
    nattn_kernel<<<dim3(100, BB * HEADS), dim3(256), smem, stream>>>(qk, rpb, out);
}
